// Encoder_76398878261379
// MI455X (gfx1250) — compile-verified, matrix operand order checked by reading, not run on hardware
//
#include <hip/hip_runtime.h>

// ---------------------------------------------------------------------------
// 3-layer GCN for MI455X (gfx1250, wave32).
//   per layer: h = x @ W (bf16 WMMA, f32 accum)
//              acc[i] = h[i]*dinv[i]^2 + b          (self-loop + bias)
//              acc[dst] += h[src]*dinv[src]*dinv[dst]  (edge scatter, f32 atomics)
//              x' = relu(acc)
// GEMM is 4.9 GFLOP total (trivial for WMMA); scatter traffic stays L2-resident
// (h and acc are 25.6MB each << 192MB L2), so we optimize for packed gathers +
// non-returning f32 atomics.
// ---------------------------------------------------------------------------

#define N_NODES 50000
#define D 128

typedef __attribute__((ext_vector_type(16))) __bf16 v16bf;
typedef __attribute__((ext_vector_type(8)))  float  v8f;

// ------------------------- small elementwise kernels -----------------------

__global__ void k_f32_to_bf16(const float* __restrict__ in,
                              __bf16* __restrict__ out, int n) {
  int i = blockIdx.x * blockDim.x + threadIdx.x;
  if (i < n) out[i] = (__bf16)in[i];
}

// W [K x N] f32 row-major  ->  WT [N x K] bf16 (so B-fragments are contiguous)
__global__ void k_w_to_bf16_t(const float* __restrict__ W,
                              __bf16* __restrict__ WT) {
  int k = blockIdx.x;   // 0..127
  int n = threadIdx.x;  // 0..127
  WT[n * D + k] = (__bf16)W[k * D + n];
}

__global__ void k_deg_init(float* __restrict__ deg, int n) {
  int i = blockIdx.x * blockDim.x + threadIdx.x;
  if (i < n) deg[i] = 1.0f;  // self-loop contributes 1 to every node's degree
}

__global__ void k_deg_edges(const long long* __restrict__ ei, int E,
                            float* __restrict__ deg) {
  int e = blockIdx.x * blockDim.x + threadIdx.x;
  if (e < E) {
    long long dst = ei[(size_t)E + e];
    atomicAdd(&deg[dst], 1.0f);
  }
}

__global__ void k_dinv(float* __restrict__ deg, int n) {
  int i = blockIdx.x * blockDim.x + threadIdx.x;
  if (i < n) {
    float d = deg[i];
    deg[i] = (d > 0.0f) ? rsqrtf(d) : 0.0f;
  }
}

// acc = h * dinv^2 (self-loop term) + bias
__global__ void k_init_acc(const float* __restrict__ H,
                           const float* __restrict__ dinv,
                           const float* __restrict__ bias,
                           float* __restrict__ acc, int total) {
  int i = blockIdx.x * blockDim.x + threadIdx.x;
  if (i < total) {
    int node = i >> 7;
    int c = i & (D - 1);
    float di = dinv[node];
    acc[i] = H[i] * di * di + bias[c];
  }
}

__global__ void k_relu(const float* __restrict__ in, float* __restrict__ out,
                       int n) {
  int i = blockIdx.x * blockDim.x + threadIdx.x;
  if (i < n) out[i] = fmaxf(in[i], 0.0f);
}

// ------------------------------- GEMM (WMMA) -------------------------------
// One wave computes a 16-row x 128-col strip of H = A(bf16) @ B(bf16).
// A: [N_NODES x D] bf16 row-major.  BT: [N x K] bf16 (W transposed).
// 16-bit A-fragment layout (ISA 7.12.2): lane holds row (lane&15), K-half
// (lane>>4): elements 0..7 = K[kb+kh*8 .. +7], 8..15 = K[kb+16+kh*8 .. +7].
// B mirrors A with columns. 3125 tiles exactly (50000/16) -> no tail, EXEC
// all-ones around every WMMA.
__global__ void __launch_bounds__(256)
k_gemm_bf16(const __bf16* __restrict__ A, const __bf16* __restrict__ BT,
            float* __restrict__ H, int ntiles) {
  int wave = (blockIdx.x * blockDim.x + threadIdx.x) >> 5;  // wave-uniform
  if (wave >= ntiles) return;
  int lane = threadIdx.x & 31;
  int m = lane & 15;       // row (A) / col (B) within tile
  int kh = lane >> 4;      // K-half select

  const __bf16* arow = A + (size_t)(wave * 16 + m) * D + kh * 8;

  v8f acc[8] = {};

#pragma unroll
  for (int kb = 0; kb < D; kb += 32) {
    v16bf a;
    ((uint4*)&a)[0] = *(const uint4*)(arow + kb);        // K = kb+kh*8 .. +7
    ((uint4*)&a)[1] = *(const uint4*)(arow + kb + 16);   // K = kb+16+kh*8 ..
#pragma unroll
    for (int t = 0; t < 8; ++t) {
      const __bf16* bp = BT + (size_t)(t * 16 + m) * D + kh * 8 + kb;
      v16bf b;
      ((uint4*)&b)[0] = *(const uint4*)(bp);
      ((uint4*)&b)[1] = *(const uint4*)(bp + 16);
      // (neg_a, A, neg_b, B, c_mod, C, reuse_a, reuse_b)
      acc[t] = __builtin_amdgcn_wmma_f32_16x16x32_bf16(
          false, a, false, b, (short)0, acc[t], false, false);
    }
  }

  // D layout: VGPR r -> row (r + 8*kh), lanes hold col = t*16 + m
  int rbase = wave * 16 + kh * 8;
#pragma unroll
  for (int t = 0; t < 8; ++t)
#pragma unroll
    for (int r = 0; r < 8; ++r)
      H[(size_t)(rbase + r) * D + t * 16 + m] = acc[t][r];
}

// ------------------------------ edge scatter -------------------------------
// One wave per edge; lane c owns channels 4c..4c+3: float4 gather of h[src],
// scale by dinv[src]*dinv[dst], 4 non-returning global f32 atomics to acc[dst].
__global__ void __launch_bounds__(256)
k_scatter(const long long* __restrict__ ei, int E,
          const float* __restrict__ H, const float* __restrict__ dinv,
          float* __restrict__ acc) {
  int t = blockIdx.x * blockDim.x + threadIdx.x;
  int e = t >> 5;
  if (e >= E) return;
  int lane = t & 31;
  long long s = ei[e];
  long long d = ei[(size_t)E + e];
  float nrm = dinv[s] * dinv[d];
  const float4 hv = *(const float4*)(H + (size_t)s * D + lane * 4);
  float* op = acc + (size_t)d * D + lane * 4;
  atomicAdd(op + 0, hv.x * nrm);
  atomicAdd(op + 1, hv.y * nrm);
  atomicAdd(op + 2, hv.z * nrm);
  atomicAdd(op + 3, hv.w * nrm);
}

// --------------------------------- driver ----------------------------------

extern "C" void kernel_launch(void* const* d_in, const int* in_sizes, int n_in,
                              void* d_out, int out_size, void* d_ws,
                              size_t ws_size, hipStream_t stream) {
  (void)n_in; (void)out_size; (void)ws_size;

  const float* x = (const float*)d_in[0];
  const long long* ei = (const long long*)d_in[1];
  const float* W[3] = {(const float*)d_in[2], (const float*)d_in[4],
                       (const float*)d_in[6]};
  const float* Bv[3] = {(const float*)d_in[3], (const float*)d_in[5],
                        (const float*)d_in[7]};
  const int E = in_sizes[1] / 2;

  // workspace partition (256B aligned): ~90 MB total
  char* p = (char*)d_ws;
  auto alloc = [&](size_t bytes) {
    char* r = p;
    p += (bytes + 255) & ~(size_t)255;
    return r;
  };
  float*  dinv = (float*)alloc((size_t)N_NODES * 4);
  __bf16* xb   = (__bf16*)alloc((size_t)N_NODES * D * 2);
  float*  h    = (float*)alloc((size_t)N_NODES * D * 4);
  float*  acc  = (float*)alloc((size_t)N_NODES * D * 4);
  float*  xbuf = (float*)alloc((size_t)N_NODES * D * 4);
  __bf16* wt[3];
  for (int l = 0; l < 3; ++l) wt[l] = (__bf16*)alloc((size_t)D * D * 2);

  const int TPB = 256;
  const int total = N_NODES * D;                 // 6.4M
  const int ntiles = N_NODES / 16;               // 3125 (exact)
  const int gN   = (N_NODES + TPB - 1) / TPB;
  const int gTot = (total + TPB - 1) / TPB;
  const int gE   = (E + TPB - 1) / TPB;
  const int gGemm = (ntiles * 32 + TPB - 1) / TPB;
  const int gScat = (int)(((size_t)E * 32 + TPB - 1) / TPB);

  // degree -> dinv (once)
  k_deg_init<<<gN, TPB, 0, stream>>>(dinv, N_NODES);
  k_deg_edges<<<gE, TPB, 0, stream>>>(ei, E, dinv);
  k_dinv<<<gN, TPB, 0, stream>>>(dinv, N_NODES);

  // weights -> bf16 transposed (once)
  for (int l = 0; l < 3; ++l)
    k_w_to_bf16_t<<<D, D, 0, stream>>>(W[l], wt[l]);

  const float* cur = x;
  for (int l = 0; l < 3; ++l) {
    k_f32_to_bf16<<<gTot, TPB, 0, stream>>>(cur, xb, total);
    k_gemm_bf16<<<gGemm, TPB, 0, stream>>>(xb, wt[l], h, ntiles);
    k_init_acc<<<gTot, TPB, 0, stream>>>(h, dinv, Bv[l], acc, total);
    k_scatter<<<gScat, TPB, 0, stream>>>(ei, E, h, dinv, acc);
    float* dst = (l == 2) ? (float*)d_out : xbuf;
    k_relu<<<gTot, TPB, 0, stream>>>(acc, dst, total);
    cur = dst;
  }
}
